// WindowAttention_20693152432804
// MI455X (gfx1250) — compile-verified, matrix operand order checked by reading, not run on hardware
//
#include <hip/hip_runtime.h>
#include <hip/hip_bf16.h>

typedef __attribute__((ext_vector_type(16))) __bf16 v16bf;
typedef __attribute__((ext_vector_type(8)))  float  v8f;

#define NTOK   64
#define DIMS   512
#define HEADS  8
#define HDIM   64
#define NWIN   2048
#define QKSCALE 0.125f   // 64^-0.5

// ---------------------------------------------------------------- helpers ---
__device__ __forceinline__ unsigned pack2bf(float a, float b) {
    union { __bf16 h[2]; unsigned u; } p;
    p.h[0] = (__bf16)a; p.h[1] = (__bf16)b;
    return p.u;
}

// A fragment (16x32 bf16) from a row-major LDS tile whose rows are 16B-aligned
// (leading dim in uint4 units). Per ISA layout the 8 dwords per lane are two
// blocks of 4 consecutive dwords -> exactly two ds_load_b128.
// Also serves as the B fragment whenever B[k][n] = src[n][k].
__device__ __forceinline__ v16bf frag_rows(const uint4* lds4, int ld4,
                                           int row0, int q0, int lane) {
    const int r = row0 + (lane & 15);
    const int half = lane >> 4;
    union { v16bf v; uint4 q[2]; } f;
    f.q[0] = lds4[r * ld4 + q0 + half];       // K 0..7   (lanes>=16: 8..15)
    f.q[1] = lds4[r * ld4 + q0 + half + 2];   // K 16..23 (lanes>=16: 24..31)
    return f.v;
}

__device__ __forceinline__ v8f wmma_bf16(v16bf a, v16bf b, v8f c) {
    return __builtin_amdgcn_wmma_f32_16x16x32_bf16(
        false, a, false, b, (short)0, c, false, false);
}

// ----------------------------------------------------------- prep kernels ---
__global__ void k_cvt_x(const float* __restrict__ x, __bf16* __restrict__ xb, int n4) {
    int t = blockIdx.x * blockDim.x + threadIdx.x;
    if (t >= n4) return;
    float4 v = ((const float4*)x)[t];
    uint2 o; o.x = pack2bf(v.x, v.y); o.y = pack2bf(v.z, v.w);
    ((uint2*)xb)[t] = o;
}

__global__ void k_cvt_wqkv(const float* __restrict__ wq, const float* __restrict__ wkv,
                           __bf16* __restrict__ wb) {
    int t = blockIdx.x * blockDim.x + threadIdx.x;
    if (t >= 1536 * DIMS) return;
    int j = t >> 9, k = t & 511;
    float w = (j < DIMS) ? wq[j * DIMS + k] : wkv[(j - DIMS) * DIMS + k];
    wb[t] = (__bf16)w;
}

__global__ void k_cvt_bqkv(const float* __restrict__ bq, const float* __restrict__ bkv,
                           float* __restrict__ bqkv) {
    int t = blockIdx.x * blockDim.x + threadIdx.x;
    if (t < 1536) bqkv[t] = (t < DIMS) ? bq[t] : bkv[t - DIMS];
}

__global__ void k_cvt_wproj(const float* __restrict__ wp, __bf16* __restrict__ wb) {
    int t = blockIdx.x * blockDim.x + threadIdx.x;
    if (t < DIMS * DIMS) wb[t] = (__bf16)wp[t];
}

__global__ void k_bias_gather(const float* __restrict__ table, float* __restrict__ biasg) {
    int t = blockIdx.x * blockDim.x + threadIdx.x;
    if (t >= HEADS * NTOK * NTOK) return;
    int h = t >> 12, ij = t & 4095, i = ij >> 6, j = ij & 63;
    int ih = i >> 3, iw = i & 7, jh = j >> 3, jw = j & 7;
    int idx = (ih - jh + 7) * 15 + (iw - jw + 7);
    biasg[t] = table[idx * HEADS + h];
}

// ------------------------------------------------------------- QKV GEMM -----
// out[b,t,0:1536] = x[b,t,:] @ Wqkv^T + bias, bf16 out. grid=(NWIN, 24)
__global__ __launch_bounds__(256)
void k_qkv_gemm(const __bf16* __restrict__ xb, const __bf16* __restrict__ wb,
                const float* __restrict__ bias, __bf16* __restrict__ qkv) {
    __shared__ __align__(16) unsigned Xs[64][36];   // rows 144B -> 16B aligned
    __shared__ __align__(16) unsigned Ws[64][36];
    uint4* Xs4 = (uint4*)&Xs[0][0];
    uint4* Ws4 = (uint4*)&Ws[0][0];
    const int b = blockIdx.x, n0g = blockIdx.y * 64;
    const int tid = threadIdx.x, lane = tid & 31, w = tid >> 5;
    const int mi = (w >> 1) * 16;
    const int nj0 = (w & 1) * 32, nj1 = nj0 + 16;
    v8f acc0 = {}, acc1 = {};
    const uint4* xu4 = (const uint4*)xb + (size_t)b * 64 * 64;  // row = 64 uint4
    const uint4* wu4 = (const uint4*)wb + (size_t)n0g * 64;
    for (int kk = 0; kk < DIMS; kk += 64) {
        const int kp4 = kk >> 3;                                 // uint4 offset
        for (int e = tid; e < 512; e += 256) {
            int r = e >> 3, c = e & 7;
            Xs4[r * 9 + c] = xu4[r * 64 + kp4 + c];
            Ws4[r * 9 + c] = wu4[r * 64 + kp4 + c];
        }
        __syncthreads();
#pragma unroll
        for (int ks = 0; ks < 2; ++ks) {
            v16bf A  = frag_rows(Xs4, 9, mi,  ks * 4, lane);
            v16bf B0 = frag_rows(Ws4, 9, nj0, ks * 4, lane);
            v16bf B1 = frag_rows(Ws4, 9, nj1, ks * 4, lane);
            acc0 = wmma_bf16(A, B0, acc0);
            acc1 = wmma_bf16(A, B1, acc1);
        }
        __syncthreads();
    }
    const int half = lane >> 4, n = lane & 15;
#pragma unroll
    for (int i = 0; i < 8; ++i) {
        int m = mi + i + half * 8;
        size_t row = ((size_t)b * 64 + m) * 1536;
        qkv[row + n0g + nj0 + n] = (__bf16)(acc0[i] + bias[n0g + nj0 + n]);
        qkv[row + n0g + nj1 + n] = (__bf16)(acc1[i] + bias[n0g + nj1 + n]);
    }
}

// -------------------------------------------------- correlation-map kernel --
// mult[b,i,j] = relu(dino_cos) * (1-clamp(sdf_cos,0,1)) * plane * cc * mm
__global__ __launch_bounds__(256)
void k_mult(const float* __restrict__ dino, const float* __restrict__ pointf,
            const float* __restrict__ sdf, const float* __restrict__ normal,
            const float* __restrict__ smask, const float* __restrict__ cmask,
            float* __restrict__ multo) {
    __shared__ __align__(16) char smem[64 * 400 * 2];   // bf16 dino (ld 400), reused f32 S
    __shared__ float pt[64][4], nr[64][4], sf[64][17], cm[64][9], sv[64], rni[64];
    unsigned short* dm = (unsigned short*)smem;          // ld 400 ushort (800B rows)
    const uint4* dm4 = (const uint4*)smem;               // ld 50 uint4
    float* Sm = (float*)smem;                            // ld 64 (reused)
    const int b = blockIdx.x, tid = threadIdx.x, lane = tid & 31, w = tid >> 5;
    const int rr = tid >> 2, q = tid & 3;                // 4 threads per row
    const size_t row = (size_t)b * 64 + rr;

    {   // row norms (dino: 96 elems/thread; sdf: 4/thread) + small features
        const float* dr = dino + row * 384;
        float s = 0.f;
        for (int i = 0; i < 96; ++i) { float v = dr[q + 4 * i]; s += v * v; }
        s += __shfl_xor(s, 1, 4);
        s += __shfl_xor(s, 2, 4);
        if (q == 0) rni[rr] = 1.f / fminf(fmaxf(sqrtf(s), 1e-20f), 1e10f);

        const float* sr = sdf + row * 16;
        float ss = 0.f;
#pragma unroll
        for (int k = 0; k < 4; ++k) { float v = sr[q * 4 + k]; ss += v * v; }
        ss += __shfl_xor(ss, 1, 4);
        ss += __shfl_xor(ss, 2, 4);
        float sninv = 1.f / fminf(fmaxf(sqrtf(ss), 1e-20f), 1e10f);
#pragma unroll
        for (int k = 0; k < 4; ++k) sf[rr][q * 4 + k] = sr[q * 4 + k] * sninv;

        if (q == 0) {
            for (int c = 0; c < 3; ++c) {
                pt[rr][c] = pointf[row * 3 + c];
                nr[rr][c] = normal[row * 3 + c];
            }
            for (int k = 0; k < 8; ++k) cm[rr][k] = cmask[row * 8 + k];
            sv[rr] = (smask[row] < 0.1f) ? 1.f : 2.f;
        }
    }
    __syncthreads();
    for (int e = tid; e < 64 * 96; e += 256) {           // normalized bf16 tile
        int r = e / 96, k4 = e - r * 96;
        float4 v = ((const float4*)(dino + ((size_t)b * 64 + r) * 384))[k4];
        float inv = rni[r];
        uint2 o; o.x = pack2bf(v.x * inv, v.y * inv); o.y = pack2bf(v.z * inv, v.w * inv);
        *(uint2*)&dm[r * 400 + 4 * k4] = o;
    }
    __syncthreads();

    const int mi = (w >> 1) * 16, nj0 = (w & 1) * 32, nj1 = nj0 + 16;
    v8f acc0 = {}, acc1 = {};
    for (int it = 0; it < 12; ++it) {                    // K = 384 = 12 * 32
        v16bf A  = frag_rows(dm4, 50, mi,  it * 4, lane);
        v16bf B0 = frag_rows(dm4, 50, nj0, it * 4, lane);
        v16bf B1 = frag_rows(dm4, 50, nj1, it * 4, lane);
        acc0 = wmma_bf16(A, B0, acc0);
        acc1 = wmma_bf16(A, B1, acc1);
    }
    __syncthreads();                                     // done reading dm
    const int half = lane >> 4, n = lane & 15;
#pragma unroll
    for (int i = 0; i < 8; ++i) {
        int m = mi + i + half * 8;
        Sm[m * 64 + nj0 + n] = fmaxf(acc0[i], 0.f);      // clip(dino, 0, inf)
        Sm[m * 64 + nj1 + n] = fmaxf(acc1[i], 0.f);
    }
    __syncthreads();
    for (int e = tid; e < 4096; e += 256) {
        int i = e >> 6, j = e & 63;
        float dv = Sm[e];
        float sd = 0.f;
        for (int k = 0; k < 16; ++k) sd += sf[i][k] * sf[j][k];
        float shadow = 1.f - fminf(fmaxf(sd, 0.f), 1.f);
        float pij = 0.f, pji = 0.f;
        for (int c = 0; c < 3; ++c) {
            float d = pt[i][c] - pt[j][c];
            pij += d * nr[i][c];
            pji -= d * nr[j][c];
        }
        float plane = __expf(-0.5f * (fabsf(pij) + fabsf(pji)));
        float cc = 0.f;
        for (int k = 0; k < 8; ++k) cc += cm[i][k] * cm[j][k];
        float mm = (sv[i] * sv[j] == 2.f) ? 1.f : 0.2f;
        multo[(size_t)b * 4096 + e] = dv * shadow * plane * cc * mm;
    }
}

// --------------------------------------------------------- attention core ---
// grid=(NWIN, HEADS): S=q@k^T, softmax(S*scale*mult+bias), O=P@v -> bf16
__global__ __launch_bounds__(256)
void k_attn(const __bf16* __restrict__ qkv, const float* __restrict__ mult,
            const float* __restrict__ biasg, __bf16* __restrict__ aout) {
    __shared__ __align__(16) unsigned Qs[64][36];
    __shared__ __align__(16) unsigned Ks[64][36];
    __shared__ __align__(16) unsigned Vt[64][36];   // V transposed: Vt[d][j]
    __shared__ __align__(16) float S[64][65];
    __shared__ __align__(16) unsigned Pb[64][36];
    uint4* Qs4 = (uint4*)&Qs[0][0];
    uint4* Ks4 = (uint4*)&Ks[0][0];
    const uint4* Vt4 = (const uint4*)&Vt[0][0];
    const uint4* Pb4 = (const uint4*)&Pb[0][0];
    unsigned short* Vts = (unsigned short*)&Vt[0][0];    // ld 72 ushort
    const int b = blockIdx.x, h = blockIdx.y;
    const int tid = threadIdx.x, lane = tid & 31, w = tid >> 5;
    const uint4* qk4 = (const uint4*)qkv + (size_t)b * 64 * 192;  // row = 192 uint4
    for (int e = tid; e < 512; e += 256) {
        int r = e >> 3, c = e & 7;
        Qs4[r * 9 + c] = qk4[r * 192 + h * 8 + c];
        Ks4[r * 9 + c] = qk4[r * 192 + 64 + h * 8 + c];
        uint4 vv = qk4[r * 192 + 128 + h * 8 + c];
        unsigned ua[4] = {vv.x, vv.y, vv.z, vv.w};
#pragma unroll
        for (int comp = 0; comp < 4; ++comp) {           // scatter-transpose V
            int cc = 4 * c + comp;                        // dword col -> d pair
            Vts[(2 * cc) * 72 + r]     = (unsigned short)(ua[comp] & 0xffff);
            Vts[(2 * cc + 1) * 72 + r] = (unsigned short)(ua[comp] >> 16);
        }
    }
    __syncthreads();

    const int mi = (w >> 1) * 16, nj0 = (w & 1) * 32, nj1 = nj0 + 16;
    const int half = lane >> 4, n = lane & 15;
    {
        v8f acc0 = {}, acc1 = {};
#pragma unroll
        for (int ks = 0; ks < 2; ++ks) {                 // K = 64
            v16bf A  = frag_rows(Qs4, 9, mi,  ks * 4, lane);
            v16bf B0 = frag_rows(Ks4, 9, nj0, ks * 4, lane);
            v16bf B1 = frag_rows(Ks4, 9, nj1, ks * 4, lane);
            acc0 = wmma_bf16(A, B0, acc0);
            acc1 = wmma_bf16(A, B1, acc1);
        }
#pragma unroll
        for (int i = 0; i < 8; ++i) {
            int m = mi + i + half * 8;
            S[m][nj0 + n] = acc0[i];
            S[m][nj1 + n] = acc1[i];
        }
    }
    __syncthreads();
    {   // softmax: 4 threads per row, 16 cols each, shuffle reductions
        const int r = tid >> 2, q = tid & 3;
        const float* mr = mult + (size_t)b * 4096 + r * 64 + q * 16;
        const float* br = biasg + (size_t)h * 4096 + r * 64 + q * 16;
        float a[16], mx = -1e30f;
#pragma unroll
        for (int jj = 0; jj < 16; ++jj) {
            a[jj] = S[r][q * 16 + jj] * QKSCALE * mr[jj] + br[jj];
            mx = fmaxf(mx, a[jj]);
        }
        mx = fmaxf(mx, __shfl_xor(mx, 1, 4));
        mx = fmaxf(mx, __shfl_xor(mx, 2, 4));
        float sum = 0.f;
#pragma unroll
        for (int jj = 0; jj < 16; ++jj) {
            a[jj] = __expf(a[jj] - mx);
            sum += a[jj];
        }
        sum += __shfl_xor(sum, 1, 4);
        sum += __shfl_xor(sum, 2, 4);
        float inv = 1.f / sum;
#pragma unroll
        for (int p = 0; p < 8; ++p)
            Pb[r][q * 8 + p] = pack2bf(a[2 * p] * inv, a[2 * p + 1] * inv);
    }
    __syncthreads();
    {
        v8f acc0 = {}, acc1 = {};
#pragma unroll
        for (int ks = 0; ks < 2; ++ks) {                 // O = P @ V (B from Vt rows)
            v16bf A  = frag_rows(Pb4, 9, mi,  ks * 4, lane);
            v16bf B0 = frag_rows(Vt4, 9, nj0, ks * 4, lane);
            v16bf B1 = frag_rows(Vt4, 9, nj1, ks * 4, lane);
            acc0 = wmma_bf16(A, B0, acc0);
            acc1 = wmma_bf16(A, B1, acc1);
        }
#pragma unroll
        for (int i = 0; i < 8; ++i) {
            int m = mi + i + half * 8;
            size_t row = ((size_t)b * 64 + m) * DIMS + h * HDIM;
            aout[row + nj0 + n] = (__bf16)acc0[i];
            aout[row + nj1 + n] = (__bf16)acc1[i];
        }
    }
}

// -------------------------------------------------------- output proj GEMM --
__global__ __launch_bounds__(256)
void k_proj(const __bf16* __restrict__ ain, const __bf16* __restrict__ wb,
            const float* __restrict__ bias, float* __restrict__ out) {
    __shared__ __align__(16) unsigned Xs[64][36];
    __shared__ __align__(16) unsigned Ws[64][36];
    uint4* Xs4 = (uint4*)&Xs[0][0];
    uint4* Ws4 = (uint4*)&Ws[0][0];
    const int b = blockIdx.x, n0g = blockIdx.y * 64;
    const int tid = threadIdx.x, lane = tid & 31, w = tid >> 5;
    const int mi = (w >> 1) * 16, nj0 = (w & 1) * 32, nj1 = nj0 + 16;
    v8f acc0 = {}, acc1 = {};
    const uint4* xu4 = (const uint4*)ain + (size_t)b * 64 * 64;
    const uint4* wu4 = (const uint4*)wb + (size_t)n0g * 64;
    for (int kk = 0; kk < DIMS; kk += 64) {
        const int kp4 = kk >> 3;
        for (int e = tid; e < 512; e += 256) {
            int r = e >> 3, c = e & 7;
            Xs4[r * 9 + c] = xu4[r * 64 + kp4 + c];
            Ws4[r * 9 + c] = wu4[r * 64 + kp4 + c];
        }
        __syncthreads();
#pragma unroll
        for (int ks = 0; ks < 2; ++ks) {
            v16bf A  = frag_rows(Xs4, 9, mi,  ks * 4, lane);
            v16bf B0 = frag_rows(Ws4, 9, nj0, ks * 4, lane);
            v16bf B1 = frag_rows(Ws4, 9, nj1, ks * 4, lane);
            acc0 = wmma_bf16(A, B0, acc0);
            acc1 = wmma_bf16(A, B1, acc1);
        }
        __syncthreads();
    }
    const int half = lane >> 4, n = lane & 15;
#pragma unroll
    for (int i = 0; i < 8; ++i) {
        int m = mi + i + half * 8;
        size_t row = ((size_t)b * 64 + m) * DIMS;
        out[row + n0g + nj0 + n] = acc0[i] + bias[n0g + nj0 + n];
        out[row + n0g + nj1 + n] = acc1[i] + bias[n0g + nj1 + n];
    }
}

// ------------------------------------------------------------------ launch --
extern "C" void kernel_launch(void* const* d_in, const int* in_sizes, int n_in,
                              void* d_out, int out_size, void* d_ws, size_t ws_size,
                              hipStream_t stream) {
    const float* x      = (const float*)d_in[0];
    const float* dino   = (const float*)d_in[1];
    const float* pointf = (const float*)d_in[2];
    // d_in[3] = color_feature (unused by reference output)
    const float* sdf    = (const float*)d_in[4];
    const float* normal = (const float*)d_in[5];
    const float* smask  = (const float*)d_in[6];
    const float* cmask  = (const float*)d_in[7];
    const float* wq     = (const float*)d_in[8];
    const float* bq     = (const float*)d_in[9];
    const float* wkv    = (const float*)d_in[10];
    const float* bkv    = (const float*)d_in[11];
    const float* rtab   = (const float*)d_in[12];
    const float* wproj  = (const float*)d_in[13];
    const float* bproj  = (const float*)d_in[14];
    float* out = (float*)d_out;

    char* ws = (char*)d_ws;
    size_t o = 0;
    __bf16* xb     = (__bf16*)(ws + o); o += (size_t)NWIN * 64 * DIMS * 2;   // 128 MB
    __bf16* wqkvb  = (__bf16*)(ws + o); o += 1536 * DIMS * 2;
    float*  bqkv   = (float*) (ws + o); o += 8192;
    __bf16* wprojb = (__bf16*)(ws + o); o += DIMS * DIMS * 2;
    __bf16* qkvb   = (__bf16*)(ws + o); o += (size_t)NWIN * 64 * 1536 * 2;   // 384 MB
    float*  multb  = (float*) (ws + o); o += (size_t)NWIN * 4096 * 4;        // 32 MB
    float*  biasg  = (float*) (ws + o); o += HEADS * 4096 * 4;
    __bf16* aoutb  = (__bf16*)(ws + o); o += (size_t)NWIN * 64 * DIMS * 2;   // 128 MB

    const int n4 = NWIN * 64 * DIMS / 4;
    k_cvt_x    <<<(n4 + 255) / 256, 256, 0, stream>>>(x, xb, n4);
    k_cvt_wqkv <<<(1536 * DIMS + 255) / 256, 256, 0, stream>>>(wq, wkv, wqkvb);
    k_cvt_bqkv <<<6, 256, 0, stream>>>(bq, bkv, bqkv);
    k_cvt_wproj<<<(DIMS * DIMS + 255) / 256, 256, 0, stream>>>(wproj, wprojb);
    k_bias_gather<<<(HEADS * 4096 + 255) / 256, 256, 0, stream>>>(rtab, biasg);

    k_qkv_gemm<<<dim3(NWIN, 24), 256, 0, stream>>>(xb, wqkvb, bqkv, qkvb);
    k_mult    <<<NWIN, 256, 0, stream>>>(dino, pointf, sdf, normal, smask, cmask, multb);
    k_attn    <<<dim3(NWIN, HEADS), 256, 0, stream>>>(qkvb, multb, biasg, aoutb);
    k_proj    <<<dim3(NWIN, 8), 256, 0, stream>>>(aoutb, wprojb, bproj, out);
}
